// CharRNN_76519137345624
// MI455X (gfx1250) — matrix operand order repaired and checked by reading, not compiled, not run on hardware
//
#include <hip/hip_runtime.h>
#include <hip/hip_bf16.h>

// CDNA5 / gfx1250 CharRNN implementation.
//
// Phases (all on `stream`, ordered by stream dependency):
//  K0: convert emb/wax/waa/wya f32 -> bf16 into d_ws
//  K1: xp[t,b,:] = emb_bf16[seq[t,b]] @ wax^T      (parallel WMMA GEMM, bf16->f32 acc)
//  K2: persistent single-workgroup recurrence:
//        h <- tanh(h @ waa^T + waa_b + xp_t), h in LDS (128 KB of the 320 KB WGP LDS),
//        32 wave32 waves, __syncthreads() per step, hs written to d_ws (bf16)
//  K3: preds = hs @ wya^T + wya_b -> d_out (f32)   (parallel WMMA GEMM)
//  K4: h_last = f32(hs[T-1])       -> d_out tail
//
// Workspace requirement: ~513 MB (bf16 weights + xp + hs).

typedef __bf16 bf16;
typedef __attribute__((ext_vector_type(16))) __bf16 v16bf;
typedef __attribute__((ext_vector_type(8)))  __bf16 v8bf;
typedef __attribute__((ext_vector_type(8)))  float  v8f;

#define T_STEPS 2048
#define BATCH   128
#define EMB     128
#define HID     512
#define NCH     256
#define MROWS   (T_STEPS * BATCH)   // 262144 "rows" of the time-flattened activations

// Assemble a 16x32 bf16 A-tile operand in contiguous K order (same as B): lane l (l<16) holds row M=l,
// K chunks [k0..k0+7] and [k0+8..k0+15]; lane l>=16 holds row M=l-16, chunks
// [k0+16..k0+23] and [k0+24..k0+31]. Caller passes the two 16-byte chunk pointers.
__device__ __forceinline__ v16bf make_a16(const bf16* p0, const bf16* p1) {
    v8bf lo = *(const v8bf*)p0;
    v8bf hi = *(const v8bf*)p1;
    v16bf r;
#pragma unroll
    for (int i = 0; i < 8; ++i) { r[i] = lo[i]; r[i + 8] = hi[i]; }
    return r;
}

__device__ __forceinline__ v16bf make_a16_lds(const bf16* p0, const bf16* p1) {
    v8bf lo = *(const v8bf*)p0;
    v8bf hi = *(const v8bf*)p1;
    v16bf r;
#pragma unroll
    for (int i = 0; i < 8; ++i) { r[i] = lo[i]; r[i + 8] = hi[i]; }
    return r;
}

// ---------------------------------------------------------------- K0: f32 -> bf16
__global__ __launch_bounds__(256) void k_convert(
    const float* __restrict__ emb, const float* __restrict__ wax,
    const float* __restrict__ waa, const float* __restrict__ wya,
    bf16* __restrict__ e, bf16* __restrict__ x,
    bf16* __restrict__ a, bf16* __restrict__ y)
{
    int i = blockIdx.x * 256 + threadIdx.x;          // grid covers 262144
    if (i < 256 * 128) e[i] = (bf16)emb[i];
    if (i < 512 * 128) x[i] = (bf16)wax[i];
    if (i < 512 * 512) a[i] = (bf16)waa[i];
    if (i < 256 * 512) y[i] = (bf16)wya[i];
}

// -------------------------------------------------- K1: embedding gather + xp GEMM
// One block = one 16-row M-tile; 8 waves x 4 col-tiles = all 512 HID columns.
__global__ __launch_bounds__(256) void k_embed_xp(
    const int*  __restrict__ seq,
    const bf16* __restrict__ embb,
    const bf16* __restrict__ waxb,
    bf16* __restrict__ xp)
{
    const int lane   = threadIdx.x & 31;
    const int wave   = threadIdx.x >> 5;
    const int l15    = lane & 15;
    const int hiHalf = lane >> 4;
    const int m0     = blockIdx.x * 16;

    const int  row  = m0 + l15;            // this lane's A-row (both lane halves)
    const int  idx  = seq[row];            // embedding gather
    const bf16* arow = embb + (size_t)idx * EMB;

    v8f acc[4];
#pragma unroll
    for (int j = 0; j < 4; ++j) acc[j] = (v8f)(0.0f);

#pragma unroll
    for (int kk = 0; kk < EMB / 32; ++kk) {
        const int ka = kk * 32 + hiHalf * 16;
        v16bf a = make_a16(arow + ka, arow + ka + 8);
        const int kb = kk * 32 + hiHalf * 16;   // B: 16 contiguous K per lane
#pragma unroll
        for (int j = 0; j < 4; ++j) {
            const int n = (wave * 4 + j) * 16 + l15;
            v16bf b = *(const v16bf*)(waxb + (size_t)n * EMB + kb);
            acc[j] = __builtin_amdgcn_wmma_f32_16x16x32_bf16(
                false, a, false, b, (short)0, acc[j], false, false);
        }
    }
    // C/D layout: VGPR r holds M = r + 8*(lane>=16), N = lane&15
#pragma unroll
    for (int j = 0; j < 4; ++j) {
        const int col = (wave * 4 + j) * 16 + l15;
#pragma unroll
        for (int r = 0; r < 8; ++r) {
            const int mrow = m0 + r + hiHalf * 8;
            xp[(size_t)mrow * HID + col] = (bf16)acc[j][r];
        }
    }
}

// ---------------------------------------------------------- K2: persistent recurrence
// Single workgroup, 1024 threads = 32 wave32 waves on one WGP.
// Wave layout: rw = wave>>2 (row-tile 0..7), cw = wave&3 (col-tiles cw*8 .. cw*8+7).
// h lives in LDS (128 KB bf16); waa streams from L2 (512 KB/step, WGP$-amortized).
__global__ __launch_bounds__(1024) void k_recurrence(
    const bf16*  __restrict__ waab,
    const float* __restrict__ waa_bias,
    const bf16*  __restrict__ xp,
    bf16*        __restrict__ hs)
{
    __shared__ bf16 hbuf[BATCH * HID];   // 128 KB of the 320 KB WGP LDS

    const int tid    = threadIdx.x;
    const int lane   = tid & 31;
    const int wave   = tid >> 5;
    const int l15    = lane & 15;
    const int hiHalf = lane >> 4;
    const int rw     = wave >> 2;
    const int cw     = wave & 3;

    for (int i = tid; i < BATCH * HID; i += 1024) hbuf[i] = (bf16)0.0f;
    __syncthreads();

    const int arow = rw * 16 + l15;      // batch row this lane reads for A

    for (int t = 0; t < T_STEPS; ++t) {
        v8f acc[8];
#pragma unroll
        for (int j = 0; j < 8; ++j) acc[j] = (v8f)(0.0f);

        for (int kk = 0; kk < HID / 32; ++kk) {
            const int ka = kk * 32 + hiHalf * 16;
            v16bf a = make_a16_lds(&hbuf[arow * HID + ka],
                                   &hbuf[arow * HID + ka + 8]);
            const int kb = kk * 32 + hiHalf * 16;
#pragma unroll
            for (int j = 0; j < 8; ++j) {        // A reused across 8 col-tiles
                const int n = (cw * 8 + j) * 16 + l15;
                v16bf b = *(const v16bf*)(waab + (size_t)n * HID + kb);
                acc[j] = __builtin_amdgcn_wmma_f32_16x16x32_bf16(
                    false, a, false, b, (short)0, acc[j], false, false);
            }
        }

        // epilogue: + bias + xp_t, tanh; publish hs[t] (global) and stage for LDS
        const size_t tbase = (size_t)t * BATCH;
#pragma unroll
        for (int j = 0; j < 8; ++j) {
            const int col  = (cw * 8 + j) * 16 + l15;
            const float bc = waa_bias[col];
#pragma unroll
            for (int r = 0; r < 8; ++r) {
                const int brow = rw * 16 + r + hiHalf * 8;
                const size_t g = (tbase + brow) * HID + col;
                float v = acc[j][r] + bc + (float)xp[g];
                v = tanhf(v);
                acc[j][r] = v;
                hs[g] = (bf16)v;
            }
        }
        __syncthreads();                 // everyone done READING old h
#pragma unroll
        for (int j = 0; j < 8; ++j) {
            const int col = (cw * 8 + j) * 16 + l15;
#pragma unroll
            for (int r = 0; r < 8; ++r) {
                const int brow = rw * 16 + r + hiHalf * 8;
                hbuf[brow * HID + col] = (bf16)acc[j][r];
            }
        }
        __syncthreads();                 // new h visible before next step
    }
}

// ------------------------------------------------------------- K3: output projection
// One block = one 16-row M-tile; 8 waves x 2 col-tiles cover NCH=256 columns; K=512.
__global__ __launch_bounds__(256) void k_out_proj(
    const bf16*  __restrict__ hs,
    const bf16*  __restrict__ wyab,
    const float* __restrict__ wya_bias,
    float* __restrict__ out)
{
    const int lane   = threadIdx.x & 31;
    const int wave   = threadIdx.x >> 5;
    const int l15    = lane & 15;
    const int hiHalf = lane >> 4;
    const int m0     = blockIdx.x * 16;

    const bf16* arow = hs + (size_t)(m0 + l15) * HID;

    v8f acc[2];
    acc[0] = (v8f)(0.0f);
    acc[1] = (v8f)(0.0f);

    for (int kk = 0; kk < HID / 32; ++kk) {
        const int ka = kk * 32 + hiHalf * 16;
        v16bf a = make_a16(arow + ka, arow + ka + 8);
        const int kb = kk * 32 + hiHalf * 16;
#pragma unroll
        for (int j = 0; j < 2; ++j) {
            const int n = (wave * 2 + j) * 16 + l15;
            v16bf b = *(const v16bf*)(wyab + (size_t)n * HID + kb);
            acc[j] = __builtin_amdgcn_wmma_f32_16x16x32_bf16(
                false, a, false, b, (short)0, acc[j], false, false);
        }
    }
#pragma unroll
    for (int j = 0; j < 2; ++j) {
        const int col  = (wave * 2 + j) * 16 + l15;
        const float bc = wya_bias[col];
#pragma unroll
        for (int r = 0; r < 8; ++r) {
            const int mrow = m0 + r + hiHalf * 8;
            out[(size_t)mrow * NCH + col] = acc[j][r] + bc;
        }
    }
}

// ------------------------------------------------------------------- K4: h_last f32
__global__ __launch_bounds__(256) void k_hlast(const bf16* __restrict__ hs,
                                               float* __restrict__ out)
{
    int i = blockIdx.x * 256 + threadIdx.x;          // 65536 elements [B,HID]
    out[i] = (float)hs[(size_t)(T_STEPS - 1) * BATCH * HID + i];
}

extern "C" void kernel_launch(void* const* d_in, const int* in_sizes, int n_in,
                              void* d_out, int out_size, void* d_ws, size_t ws_size,
                              hipStream_t stream) {
    // setup_inputs() order: input_seq, emb, waa_w, waa_b, wax_w, wya_w, wya_b
    const int*   seq    = (const int*)d_in[0];
    const float* emb    = (const float*)d_in[1];
    const float* waa_w  = (const float*)d_in[2];
    const float* waa_b  = (const float*)d_in[3];
    const float* wax_w  = (const float*)d_in[4];
    const float* wya_w  = (const float*)d_in[5];
    const float* wya_b  = (const float*)d_in[6];
    float* out = (float*)d_out;

    // workspace layout (bf16 elements); total ~513 MB
    bf16* e  = (bf16*)d_ws;
    bf16* x  = e  + 256 * 128;
    bf16* a  = x  + 512 * 128;
    bf16* y  = a  + 512 * 512;
    bf16* xp = y  + 256 * 512;
    bf16* hs = xp + (size_t)MROWS * HID;

    k_convert   <<<1024,       256,  0, stream>>>(emb, wax_w, waa_w, wya_w, e, x, a, y);
    k_embed_xp  <<<MROWS / 16, 256,  0, stream>>>(seq, e, x, xp);
    k_recurrence<<<1,          1024, 0, stream>>>(a, waa_b, xp, hs);
    k_out_proj  <<<MROWS / 16, 256,  0, stream>>>(hs, y, wya_b, out);
    k_hlast     <<<256,        256,  0, stream>>>(hs, out + (size_t)MROWS * NCH);
}
